// RNNModel_71803263255198
// MI455X (gfx1250) — compile-verified, matrix operand order checked by reading, not run on hardware
//
#include <hip/hip_runtime.h>

#define S_LEN 256
#define BATCH 16
#define NHID 1024
#define NGATE 3072
#define NTOK 32000
#define ROWS (S_LEN * BATCH)   // 4096

typedef __attribute__((ext_vector_type(16))) __bf16          v16bf;
typedef __attribute__((ext_vector_type(16))) unsigned short  v16u16;
typedef __attribute__((ext_vector_type(8)))  unsigned short  u16x8;
typedef __attribute__((ext_vector_type(8)))  float           v8f;

// ---------- helpers ----------

__device__ __forceinline__ unsigned short f2bf(float f) {
    unsigned int u = __builtin_bit_cast(unsigned int, f);
    unsigned int r = u + 0x7FFFu + ((u >> 16) & 1u);   // round-to-nearest-even
    return (unsigned short)(r >> 16);
}

// Load one 16x32 bf16 WMMA fragment row (A) or column (B==W row) for this lane.
// p points at element [row, kbase] of a K-contiguous row; elements 0..7 come
// from kbase..kbase+7, elements 8..15 from kbase+16..kbase+23 (ISA 7.12.2).
// Works for global or LDS (generic pointer -> global_load_b128 / ds_load_b128).
__device__ __forceinline__ v16bf load_frag16(const unsigned short* p) {
    u16x8 lo = *(const u16x8*)p;
    u16x8 hi = *(const u16x8*)(p + 16);
    v16u16 v;
#pragma unroll
    for (int i = 0; i < 8; ++i) { v[i] = lo[i]; v[i + 8] = hi[i]; }
    return __builtin_bit_cast(v16bf, v);
}

// Low 32 bits of a generic pointer to LDS == LDS byte offset (ISA §10.2:
// the LDS aperture occupies ADDR[63:32], ADDR[31:0] is the LDS address).
__device__ __forceinline__ unsigned lds_addr32(const void* p) {
    return (unsigned)(unsigned long long)p;
}

// ---------- precision conversion ----------

__global__ void f32_to_bf16_kernel(const float* __restrict__ src,
                                   unsigned short* __restrict__ dst, long n) {
    long i = (long)blockIdx.x * blockDim.x + threadIdx.x;
    long stride = (long)gridDim.x * blockDim.x;
    for (; i < n; i += stride) dst[i] = f2bf(src[i]);
}

// gather emb[tokens[s,b]] -> bf16 [s*B+b, 1024]; flip reverses time for right stream
__global__ void gather_embed_kernel(const int* __restrict__ tokens,
                                    const float* __restrict__ emb,
                                    unsigned short* __restrict__ out, int flip) {
    int sb = blockIdx.x;              // 0..4095
    int s = sb >> 4, b = sb & 15;
    int srow = flip ? (S_LEN - 1 - s) : s;
    int tok = tokens[srow * BATCH + b];
    const float* src = emb + (long)tok * NHID;
    unsigned short* dst = out + (long)sb * NHID;
    for (int j = threadIdx.x; j < NHID; j += blockDim.x) dst[j] = f2bf(src[j]);
}

// ---------- generic WMMA GEMM:  C[M,N] = A[M,K] @ W[N,K]^T (+ bias[N]) ----------
// block: 256 thr = 8 waves (2x4), block tile 64(M) x 128(N), 32x32 per wave.
// Per k-step the 128x32 W slice (8KB) is staged into LDS with
// GLOBAL_LOAD_ASYNC_TO_LDS_B128 (ASYNCcnt); B fragments then come from LDS,
// A fragments straight from global (4x reuse served by L0 on same lines).

__global__ __launch_bounds__(256) void gemm_bf16_wmma_kernel(
    const unsigned short* __restrict__ A, int lda,
    const unsigned short* __restrict__ W, int ldw,
    const float* __restrict__ bias,
    float* __restrict__ C, int ldc, int K) {
    __shared__ __attribute__((aligned(128))) unsigned short Wlds[128 * 32]; // 8KB

    int lane = threadIdx.x & 31;
    int wave = threadIdx.x >> 5;
    int l16  = lane & 15;
    int half = (lane >> 4) & 1;
    int tileM  = blockIdx.y * 64 + (wave >> 2) * 32;
    int tileNb = blockIdx.x * 128;
    int wn     = wave & 3;

    unsigned ldsbase = lds_addr32(Wlds);

    v8f acc[2][2] = {};
    for (int k0 = 0; k0 < K; k0 += 32) {
        __syncthreads();   // previous W slice fully consumed by all waves
        // 512 x 16B chunks: thread t copies chunks t and t+256
#pragma unroll
        for (int i = 0; i < 2; ++i) {
            int c = threadIdx.x + i * 256;
            int row = c >> 2, q = c & 3;
            unsigned long long src = (unsigned long long)(const void*)(
                W + (long)(tileNb + row) * ldw + k0 + q * 8);
            unsigned dst = ldsbase + (unsigned)(row * 64 + q * 16);
            asm volatile("global_load_async_to_lds_b128 %0, %1, off"
                         :: "v"(dst), "v"(src) : "memory");
        }
        if (k0 + 128 < K)   // pull upcoming A lines toward L2 (global_prefetch_b8)
            __builtin_prefetch((const void*)(A + (long)(tileM + l16) * lda + k0 + 128), 0, 1);
        asm volatile("s_wait_asynccnt 0x0" ::: "memory");
        __syncthreads();   // staged slice visible to whole block

        int kb = half * 8;
        v16bf a[2], b[2];
#pragma unroll
        for (int mi = 0; mi < 2; ++mi)
            a[mi] = load_frag16(A + (long)(tileM + mi * 16 + l16) * lda + k0 + kb);
#pragma unroll
        for (int ni = 0; ni < 2; ++ni)
            b[ni] = load_frag16(Wlds + (wn * 32 + ni * 16 + l16) * 32 + kb);
#pragma unroll
        for (int mi = 0; mi < 2; ++mi)
#pragma unroll
            for (int ni = 0; ni < 2; ++ni)
                acc[mi][ni] = __builtin_amdgcn_wmma_f32_16x16x32_bf16(
                    false, a[mi], false, b[ni], (short)0, acc[mi][ni], false, false);
    }
#pragma unroll
    for (int mi = 0; mi < 2; ++mi)
#pragma unroll
        for (int ni = 0; ni < 2; ++ni) {
            int n = tileNb + wn * 32 + ni * 16 + l16;
            float bv = bias ? bias[n] : 0.0f;
#pragma unroll
            for (int g = 0; g < 8; ++g) {
                int m = tileM + mi * 16 + g + half * 8;   // C/D layout ISA 7.12.2
                C[(long)m * ldc + n] = acc[mi][ni][g] + bv;
            }
        }
}

// ---------- sequential GRU (1 workgroup per stream, persistent over S) ----------
// LDS: h_bf 32KB + h_f 64KB + gh 192KB = 288KB  (CDNA5 WGP has 320KB)

__global__ __launch_bounds__(1024) void gru_seq_kernel(
    const float* __restrict__ xgL, const float* __restrict__ xgR,
    const unsigned short* __restrict__ WhhL, const unsigned short* __restrict__ WhhR,
    const float* __restrict__ bhhL, const float* __restrict__ bhhR,
    const float* __restrict__ h0L, const float* __restrict__ h0R,
    float* __restrict__ outfL, float* __restrict__ outfR,
    unsigned short* __restrict__ outbL, unsigned short* __restrict__ outbR) {
    extern __shared__ char smem[];
    unsigned short* h_bf = (unsigned short*)smem;                              // [16,1024]
    float* h_f = (float*)(smem + BATCH * NHID * 2);                            // [16,1024]
    float* gh  = (float*)(smem + BATCH * NHID * 2 + BATCH * NHID * 4);         // [16,3072]

    bool right = blockIdx.x != 0;
    const float* xg = right ? xgR : xgL;
    const unsigned short* Whh = right ? WhhR : WhhL;
    const float* bhh = right ? bhhR : bhhL;
    const float* h0  = right ? h0R : h0L;
    float* outf = right ? outfR : outfL;
    unsigned short* outb = right ? outbR : outbL;

    for (int i = threadIdx.x; i < BATCH * NHID; i += blockDim.x) {
        float v = h0[i];
        h_f[i] = v;
        h_bf[i] = f2bf(v);
    }
    __syncthreads();

    int lane = threadIdx.x & 31;
    int wave = threadIdx.x >> 5;          // 0..31
    int l16  = lane & 15;
    int half = (lane >> 4) & 1;

    for (int s = 0; s < S_LEN; ++s) {
        // gh[16,3072] = h[16,1024] @ Whh^T : 192 N-tiles, 6 per wave
#pragma unroll 1
        for (int t = 0; t < 6; ++t) {
            int n0 = (wave * 6 + t) * 16;
            v8f acc = {};
            for (int k0 = 0; k0 < NHID; k0 += 32) {
                int kb = k0 + half * 8;
                v16bf a = load_frag16(h_bf + l16 * NHID + kb);
                v16bf b = load_frag16(Whh + (long)(n0 + l16) * NHID + kb);
                acc = __builtin_amdgcn_wmma_f32_16x16x32_bf16(
                    false, a, false, b, (short)0, acc, false, false);
            }
#pragma unroll
            for (int g = 0; g < 8; ++g)
                gh[(g + half * 8) * NGATE + n0 + l16] = acc[g];
        }
        __syncthreads();

        const float* xgs = xg + (long)s * BATCH * NGATE;
        for (int i = threadIdx.x; i < BATCH * NHID; i += blockDim.x) {
            int b = i >> 10, j = i & (NHID - 1);
            float xr = xgs[b * NGATE + j];
            float xz = xgs[b * NGATE + NHID + j];
            float xn = xgs[b * NGATE + 2 * NHID + j];
            float hr = gh[b * NGATE + j] + bhh[j];
            float hz = gh[b * NGATE + NHID + j] + bhh[NHID + j];
            float hn = gh[b * NGATE + 2 * NHID + j] + bhh[2 * NHID + j];
            float r = 1.0f / (1.0f + __expf(-(xr + hr)));
            float z = 1.0f / (1.0f + __expf(-(xz + hz)));
            float nn = tanhf(xn + r * hn);
            float hnew = (1.0f - z) * nn + z * h_f[i];
            h_f[i] = hnew;
            unsigned short hb = f2bf(hnew);
            h_bf[i] = hb;
            outf[(long)s * BATCH * NHID + i] = hnew;
            outb[(long)s * BATCH * NHID + i] = hb;
        }
        __syncthreads();
    }
}

// ---------- attention score: sc[row] = tanh(proj[row,:]) . v + bv ----------

__global__ __launch_bounds__(256) void attn_score_kernel(
    const float* __restrict__ proj, const float* __restrict__ v,
    const float* __restrict__ bv, float* __restrict__ sc) {
    int row = blockIdx.x * 8 + (threadIdx.x >> 5);
    int lane = threadIdx.x & 31;
    const float* p = proj + (long)row * NHID;
    float sum = 0.f;
    for (int j = lane; j < NHID; j += 32) sum += tanhf(p[j]) * v[j];
#pragma unroll
    for (int off = 16; off > 0; off >>= 1) sum += __shfl_down(sum, off, 32);
    if (lane == 0) sc[row] = sum + bv[0];
}

// ---------- per-batch max/exp/prefix-sum of scores (tiny) ----------

__global__ void softmax_prep_kernel(const float* __restrict__ scL, const float* __restrict__ scR,
                                    float* __restrict__ eL, float* __restrict__ eR,
                                    float* __restrict__ cumL, float* __restrict__ cumR) {
    int t = threadIdx.x;
    if (t >= 32) return;
    const float* sc = (t < 16) ? scL : scR;
    float* e   = (t < 16) ? eL : eR;
    float* cum = (t < 16) ? cumL : cumR;
    int b = t & 15;
    float m = -1e30f;
    for (int s = 0; s < S_LEN; ++s) m = fmaxf(m, sc[s * BATCH + b]);
    float run = 0.f;
    for (int s = 0; s < S_LEN; ++s) {
        float ev = __expf(sc[s * BATCH + b] - m);
        run += ev;
        e[s * BATCH + b] = ev;
        cum[s * BATCH + b] = run;
    }
}

// ---------- context cumsum -> bf16 context [S,B,2H] (right stream reversed) ----------

__global__ __launch_bounds__(128) void context_kernel(
    const float* __restrict__ eL, const float* __restrict__ cumL, const float* __restrict__ outfL,
    const float* __restrict__ eR, const float* __restrict__ cumR, const float* __restrict__ outfR,
    unsigned short* __restrict__ ctx_bf) {
    int bx = blockIdx.x;              // 0..255
    int st = bx >> 7;                 // 0=left, 1=right(flipped order)
    int rem = bx & 127;
    int b = rem >> 3;
    int h = (rem & 7) * 128 + threadIdx.x;
    const float* e    = st ? eR : eL;
    const float* cum  = st ? cumR : cumL;
    const float* outf = st ? outfR : outfL;
    float num = 0.f;
    for (int s = 0; s < S_LEN; ++s) {
        int rb = s * BATCH + b;
        num += e[rb] * outf[(long)rb * NHID + h];
        float val = num / cum[rb];
        int sw = st ? (S_LEN - 1 - s) : s;   // suffix softmax == prefix in flipped coords
        ctx_bf[((long)(sw * BATCH + b)) * (2 * NHID) + st * NHID + h] = f2bf(val);
    }
}

// ---------- host ----------

extern "C" void kernel_launch(void* const* d_in, const int* in_sizes, int n_in,
                              void* d_out, int out_size, void* d_ws, size_t ws_size,
                              hipStream_t stream) {
    const int*   data_left  = (const int*)d_in[0];
    const int*   data_right = (const int*)d_in[1];
    const float* hidden_l   = (const float*)d_in[2];
    const float* hidden_r   = (const float*)d_in[3];
    const float* emb        = (const float*)d_in[4];
    const float* Wih_l = (const float*)d_in[5],  *Whh_l = (const float*)d_in[6];
    const float* bih_l = (const float*)d_in[7],  *bhh_l = (const float*)d_in[8];
    const float* Wih_r = (const float*)d_in[9],  *Whh_r = (const float*)d_in[10];
    const float* bih_r = (const float*)d_in[11], *bhh_r = (const float*)d_in[12];
    const float* Wattn_l = (const float*)d_in[13], *battn_l = (const float*)d_in[14];
    const float* v_l = (const float*)d_in[15], *bv_l = (const float*)d_in[16];
    const float* Wattn_r = (const float*)d_in[17], *battn_r = (const float*)d_in[18];
    const float* v_r = (const float*)d_in[19], *bv_r = (const float*)d_in[20];
    const float* Wdec = (const float*)d_in[21], *bdec = (const float*)d_in[22];

    // workspace bump allocator (256B aligned)
    char* ws = (char*)d_ws;
    size_t off = 0;
    auto alloc = [&](size_t bytes) -> char* {
        char* p = ws + off;
        off += (bytes + 255) & ~(size_t)255;
        return p;
    };
    unsigned short* xembL = (unsigned short*)alloc((size_t)ROWS * NHID * 2);
    unsigned short* xembR = (unsigned short*)alloc((size_t)ROWS * NHID * 2);
    unsigned short* WihLb = (unsigned short*)alloc((size_t)NGATE * NHID * 2);
    unsigned short* WihRb = (unsigned short*)alloc((size_t)NGATE * NHID * 2);
    unsigned short* WhhLb = (unsigned short*)alloc((size_t)NGATE * NHID * 2);
    unsigned short* WhhRb = (unsigned short*)alloc((size_t)NGATE * NHID * 2);
    unsigned short* WatLb = (unsigned short*)alloc((size_t)NHID * NHID * 2);
    unsigned short* WatRb = (unsigned short*)alloc((size_t)NHID * NHID * 2);
    unsigned short* Wdecb = (unsigned short*)alloc((size_t)NTOK * 2 * NHID * 2);
    float* xgL   = (float*)alloc((size_t)ROWS * NGATE * 4);
    float* xgR   = (float*)alloc((size_t)ROWS * NGATE * 4);
    float* outfL = (float*)alloc((size_t)ROWS * NHID * 4);
    float* outfR = (float*)alloc((size_t)ROWS * NHID * 4);
    unsigned short* outbL = (unsigned short*)alloc((size_t)ROWS * NHID * 2);
    unsigned short* outbR = (unsigned short*)alloc((size_t)ROWS * NHID * 2);
    float* projL = (float*)alloc((size_t)ROWS * NHID * 4);
    float* projR = (float*)alloc((size_t)ROWS * NHID * 4);
    float* scL  = (float*)alloc((size_t)ROWS * 4);
    float* scR  = (float*)alloc((size_t)ROWS * 4);
    float* eL   = (float*)alloc((size_t)ROWS * 4);
    float* eR   = (float*)alloc((size_t)ROWS * 4);
    float* cumL = (float*)alloc((size_t)ROWS * 4);
    float* cumR = (float*)alloc((size_t)ROWS * 4);
    unsigned short* ctxb = (unsigned short*)alloc((size_t)ROWS * 2 * NHID * 2);
    (void)ws_size; (void)n_in; (void)in_sizes; (void)out_size;

    auto cvt = [&](const float* src, unsigned short* dst, long n) {
        int blocks = (int)((n + 1023) / 1024);
        f32_to_bf16_kernel<<<blocks, 256, 0, stream>>>(src, dst, n);
    };
    cvt(Wih_l, WihLb, (long)NGATE * NHID);
    cvt(Wih_r, WihRb, (long)NGATE * NHID);
    cvt(Whh_l, WhhLb, (long)NGATE * NHID);
    cvt(Whh_r, WhhRb, (long)NGATE * NHID);
    cvt(Wattn_l, WatLb, (long)NHID * NHID);
    cvt(Wattn_r, WatRb, (long)NHID * NHID);
    cvt(Wdec, Wdecb, (long)NTOK * 2 * NHID);

    gather_embed_kernel<<<ROWS, 256, 0, stream>>>(data_left, emb, xembL, 0);
    gather_embed_kernel<<<ROWS, 256, 0, stream>>>(data_right, emb, xembR, 1);

    // xg = xemb @ Wih^T + bih     [4096,3072]
    gemm_bf16_wmma_kernel<<<dim3(NGATE / 128, ROWS / 64), 256, 0, stream>>>(
        xembL, NHID, WihLb, NHID, bih_l, xgL, NGATE, NHID);
    gemm_bf16_wmma_kernel<<<dim3(NGATE / 128, ROWS / 64), 256, 0, stream>>>(
        xembR, NHID, WihRb, NHID, bih_r, xgR, NGATE, NHID);

    // sequential GRU, both streams (288KB dynamic LDS per workgroup; CDNA5 WGP = 320KB)
    size_t gru_smem = (size_t)BATCH * NHID * 2 + (size_t)BATCH * NHID * 4 +
                      (size_t)BATCH * NGATE * 4;
    hipFuncSetAttribute((const void*)gru_seq_kernel,
                        hipFuncAttributeMaxDynamicSharedMemorySize, (int)gru_smem);
    gru_seq_kernel<<<2, 1024, gru_smem, stream>>>(
        xgL, xgR, WhhLb, WhhRb, bhh_l, bhh_r, hidden_l, hidden_r,
        outfL, outfR, outbL, outbR);

    // attention projection: proj = out @ Wattn^T + battn   [4096,1024]
    gemm_bf16_wmma_kernel<<<dim3(NHID / 128, ROWS / 64), 256, 0, stream>>>(
        outbL, NHID, WatLb, NHID, battn_l, projL, NHID, NHID);
    gemm_bf16_wmma_kernel<<<dim3(NHID / 128, ROWS / 64), 256, 0, stream>>>(
        outbR, NHID, WatRb, NHID, battn_r, projR, NHID, NHID);

    attn_score_kernel<<<ROWS / 8, 256, 0, stream>>>(projL, v_l, bv_l, scL);
    attn_score_kernel<<<ROWS / 8, 256, 0, stream>>>(projR, v_r, bv_r, scR);

    softmax_prep_kernel<<<1, 64, 0, stream>>>(scL, scR, eL, eR, cumL, cumR);

    context_kernel<<<256, 128, 0, stream>>>(eL, cumL, outfL, eR, cumR, outfR, ctxb);

    // decoder: out = ctx @ Wdec^T + bdec   [4096,32000]
    gemm_bf16_wmma_kernel<<<dim3(NTOK / 128, ROWS / 64), 256, 0, stream>>>(
        ctxb, 2 * NHID, Wdecb, 2 * NHID, bdec, (float*)d_out, NTOK, 2 * NHID);
}
